// MLP_52493090292303
// MI455X (gfx1250) — matrix operand order repaired and checked by reading, not compiled, not run on hardware
//
#include <hip/hip_runtime.h>

typedef __attribute__((ext_vector_type(16))) _Float16 v16h;
typedef __attribute__((ext_vector_type(8)))  _Float16 v8h;
typedef __attribute__((ext_vector_type(8)))  float    v8f;

#define SLOPE 0.01f
#define EPS   1e-5f

// ---------------------------------------------------------------------------
// Shared epilogue: per-wave 16x128 f32 tile -> LDS transpose -> coalesced f16
// store, plus deterministic per-block column sum / sumsq partials.
// C/D WMMA layout: lane l (<16): col n=l, rows 0..7 in v[0..7]; lane l+16:
// col n=l, rows 8..15.
// ---------------------------------------------------------------------------
__device__ __forceinline__ void tile_epilogue(
    v8f (&acc)[8], int wave, int lane, int tid, int m0,
    _Float16* __restrict__ hout, float* __restrict__ partials, int blk,
    _Float16 (&stage)[8][16][136], float (&sS)[16][128], float (&sQ)[16][128])
{
    const int rbase = (lane < 16) ? 0 : 8;
    const int hslot = wave * 2 + ((lane < 16) ? 0 : 1);
#pragma unroll
    for (int nt = 0; nt < 8; ++nt) {
        const int n = nt * 16 + (lane & 15);
        float s1 = 0.f, s2 = 0.f;
#pragma unroll
        for (int r = 0; r < 8; ++r) {
            float v = acc[nt][r];
            stage[wave][rbase + r][n] = (_Float16)v;
            s1 += v;
            s2 += v * v;
        }
        sS[hslot][n] = s1;
        sQ[hslot][n] = s2;
    }
    __syncthreads();

    // coalesced store: wave writes its 16 rows x 128 cols as b128 chunks
#pragma unroll
    for (int t = 0; t < 8; ++t) {
        int c   = lane + 32 * t;          // 0..255 chunk id (16 rows x 16 chunks)
        int row = c >> 4;
        int off = (c & 15) << 3;
        v8h d = *(const v8h*)&stage[wave][row][off];
        *(v8h*)(hout + (size_t)(m0 + row) * 128 + off) = d;
    }

    // deterministic fixed-order block reduction of stats
    {
        int f = tid & 127;
        float s = 0.f;
        if (tid < 128) {
#pragma unroll
            for (int w = 0; w < 16; ++w) s += sS[w][f];
        } else {
#pragma unroll
            for (int w = 0; w < 16; ++w) s += sQ[w][f];
        }
        partials[(size_t)blk * 256 + tid] = s;
    }
}

// ---------------------------------------------------------------------------
// Prep: convert weights to f16, zero-padding W1 (K 9->32) and W6 (N 9->16).
// ---------------------------------------------------------------------------
__global__ void prep_weights(const float* __restrict__ W1, const float* __restrict__ W2,
                             const float* __restrict__ W3, const float* __restrict__ W4,
                             const float* __restrict__ W5, const float* __restrict__ W6,
                             _Float16* __restrict__ wf)
{
    int i = blockIdx.x * blockDim.x + threadIdx.x;
    if (i < 4096) {                              // W1p [128][32]
        int n = i >> 5, k = i & 31;
        wf[i] = (_Float16)((k < 9) ? W1[n * 9 + k] : 0.f);
    } else if (i < 69632) {                      // W2..W5 [128][128] each
        int j = i - 4096;
        int L = j >> 14, e = j & 16383;
        const float* Ws = (L == 0) ? W2 : (L == 1) ? W3 : (L == 2) ? W4 : W5;
        wf[i] = (_Float16)Ws[e];
    } else if (i < 71680) {                      // W6p [16][128]
        int j = i - 69632;
        int n = j >> 7, k = j & 127;
        wf[i] = (_Float16)((n < 9) ? W6[n * 128 + k] : 0.f);
    }
}

// ---------------------------------------------------------------------------
// Layer 1: x[B][9] f32 -> h1[B][128] f16 (raw pre-BN) + stats partials.
// ---------------------------------------------------------------------------
__global__ void __launch_bounds__(256) layer_first(
    const float* __restrict__ x, const _Float16* __restrict__ W1p,
    const float* __restrict__ b1, _Float16* __restrict__ hout,
    float* __restrict__ partials)
{
    __shared__ _Float16 stage[8][16][136];
    __shared__ float sS[16][128];
    __shared__ float sQ[16][128];

    const int tid = threadIdx.x, wave = tid >> 5, lane = tid & 31;
    const int m0 = blockIdx.x * 128 + wave * 16;
    const int mrow = m0 + (lane & 15);

    // A fragment (16x32, K padded 9->32): lanes<16 hold K{0..7,16..23},
    // lanes>=16 hold K{8..15,24..31}; only K<9 nonzero.
    v16h af;
#pragma unroll
    for (int j = 0; j < 16; ++j) af[j] = (_Float16)0.f;
    const float* xr = x + (size_t)mrow * 9;
    if (lane < 16) {
#pragma unroll
        for (int j = 0; j < 8; ++j) af[j] = (_Float16)xr[j];   // K=0..7
    } else {
        af[0] = (_Float16)xr[8];                               // K=8
    }

    v8f acc[8];
#pragma unroll
    for (int nt = 0; nt < 8; ++nt) {
        float bv = b1[nt * 16 + (lane & 15)];
#pragma unroll
        for (int r = 0; r < 8; ++r) acc[nt][r] = bv;
    }

    const int wk = (lane < 16) ? 0 : 8;
#pragma unroll
    for (int nt = 0; nt < 8; ++nt) {
        int n = nt * 16 + (lane & 15);
        v8h b0 = *(const v8h*)(W1p + (size_t)n * 32 + wk);
        v8h b1 = *(const v8h*)(W1p + (size_t)n * 32 + wk + 16);
        v16h bf = __builtin_shufflevector(b0, b1, 0, 1, 2, 3, 4, 5, 6, 7, 8, 9, 10, 11, 12, 13, 14, 15);
        acc[nt] = __builtin_amdgcn_wmma_f32_16x16x32_f16(
            false, af, false, bf, (short)0, acc[nt], false, false);
    }

    tile_epilogue(acc, wave, lane, tid, m0, hout, partials, blockIdx.x,
                  stage, sS, sQ);
}

// ---------------------------------------------------------------------------
// Finalize BN stats for one layer: deterministic reduction over block partials.
// ---------------------------------------------------------------------------
__global__ void finalize_bn(const float* __restrict__ P, int nblocks,
                            const float* __restrict__ g, const float* __restrict__ be,
                            float* __restrict__ sc, float* __restrict__ sh, float invB)
{
    int f = threadIdx.x;   // 128 threads
    float S = 0.f, Q = 0.f;
    for (int b = 0; b < nblocks; ++b) {
        S += P[(size_t)b * 256 + f];
        Q += P[(size_t)b * 256 + 128 + f];
    }
    float mu  = S * invB;
    float var = Q * invB - mu * mu;
    float rs  = rsqrtf(var + EPS);
    float scv = g[f] * rs;
    sc[f] = scv;
    sh[f] = be[f] - mu * scv;
}

// ---------------------------------------------------------------------------
// Middle layers 2..5: h_prev (raw f16) --BN affine+lrelu--> A frags,
// GEMM with W (f16), write raw h (f16) + stats partials.
// ---------------------------------------------------------------------------
__global__ void __launch_bounds__(256) layer_mid(
    const _Float16* __restrict__ hin, const float* __restrict__ sc,
    const float* __restrict__ sh, const _Float16* __restrict__ Wf,
    const float* __restrict__ bias, _Float16* __restrict__ hout,
    float* __restrict__ partials)
{
    __shared__ _Float16 stage[8][16][136];
    __shared__ float sS[16][128];
    __shared__ float sQ[16][128];
    __shared__ float s_sc[128];
    __shared__ float s_sh[128];

    const int tid = threadIdx.x, wave = tid >> 5, lane = tid & 31;
    if (tid < 128) { s_sc[tid] = sc[tid]; s_sh[tid] = sh[tid]; }
    __syncthreads();

    const int m0 = blockIdx.x * 128 + wave * 16;
    const int mrow = m0 + (lane & 15);
    const int koff = (lane < 16) ? 0 : 8;

    v8f acc[8];
#pragma unroll
    for (int nt = 0; nt < 8; ++nt) {
        float bv = bias[nt * 16 + (lane & 15)];
#pragma unroll
        for (int r = 0; r < 8; ++r) acc[nt][r] = bv;
    }

    const _Float16* hrow = hin + (size_t)mrow * 128;

#pragma unroll
    for (int kt = 0; kt < 4; ++kt) {
        const int kb = kt * 32 + koff;
        v8h h0 = *(const v8h*)(hrow + kb);        // K = kb .. kb+7
        v8h h1 = *(const v8h*)(hrow + kb + 16);   // K = kb+16 .. kb+23
        v16h af;
#pragma unroll
        for (int j = 0; j < 8; ++j) {
            float v0 = (float)h0[j] * s_sc[kb + j] + s_sh[kb + j];
            v0 = fmaxf(v0, SLOPE * v0);           // leaky relu
            af[j] = (_Float16)v0;
            float v1 = (float)h1[j] * s_sc[kb + 16 + j] + s_sh[kb + 16 + j];
            v1 = fmaxf(v1, SLOPE * v1);
            af[j + 8] = (_Float16)v1;
        }
        const int wk = kt * 32 + ((lane < 16) ? 0 : 8);
#pragma unroll
        for (int nt = 0; nt < 8; ++nt) {
            int n = nt * 16 + (lane & 15);
            v8h b0 = *(const v8h*)(Wf + (size_t)n * 128 + wk);
            v8h b1 = *(const v8h*)(Wf + (size_t)n * 128 + wk + 16);
            v16h bf = __builtin_shufflevector(b0, b1, 0, 1, 2, 3, 4, 5, 6, 7, 8, 9, 10, 11, 12, 13, 14, 15);
            acc[nt] = __builtin_amdgcn_wmma_f32_16x16x32_f16(
                false, af, false, bf, (short)0, acc[nt], false, false);
        }
    }

    tile_epilogue(acc, wave, lane, tid, m0, hout, partials, blockIdx.x,
                  stage, sS, sQ);
}

// ---------------------------------------------------------------------------
// Layer 6: BN5 affine + lrelu on h5, GEMM with W6 (N padded 9->16),
// write f32 out [B][9].
// ---------------------------------------------------------------------------
__global__ void __launch_bounds__(256) layer_last(
    const _Float16* __restrict__ hin, const float* __restrict__ sc,
    const float* __restrict__ sh, const _Float16* __restrict__ W6p,
    const float* __restrict__ b6, float* __restrict__ out)
{
    __shared__ float s_sc[128];
    __shared__ float s_sh[128];

    const int tid = threadIdx.x, wave = tid >> 5, lane = tid & 31;
    if (tid < 128) { s_sc[tid] = sc[tid]; s_sh[tid] = sh[tid]; }
    __syncthreads();

    const int m0 = blockIdx.x * 128 + wave * 16;
    const int mrow = m0 + (lane & 15);
    const int koff = (lane < 16) ? 0 : 8;
    const int n = lane & 15;

    float bv = (n < 9) ? b6[n] : 0.f;
    v8f acc;
#pragma unroll
    for (int r = 0; r < 8; ++r) acc[r] = bv;

    const _Float16* hrow = hin + (size_t)mrow * 128;

#pragma unroll
    for (int kt = 0; kt < 4; ++kt) {
        const int kb = kt * 32 + koff;
        v8h h0 = *(const v8h*)(hrow + kb);
        v8h h1 = *(const v8h*)(hrow + kb + 16);
        v16h af;
#pragma unroll
        for (int j = 0; j < 8; ++j) {
            float v0 = (float)h0[j] * s_sc[kb + j] + s_sh[kb + j];
            v0 = fmaxf(v0, SLOPE * v0);
            af[j] = (_Float16)v0;
            float v1 = (float)h1[j] * s_sc[kb + 16 + j] + s_sh[kb + 16 + j];
            v1 = fmaxf(v1, SLOPE * v1);
            af[j + 8] = (_Float16)v1;
        }
        const int wk = kt * 32 + ((lane < 16) ? 0 : 8);
        v8h b0 = *(const v8h*)(W6p + (size_t)n * 128 + wk);
        v8h b1 = *(const v8h*)(W6p + (size_t)n * 128 + wk + 16);
        v16h bf = __builtin_shufflevector(b0, b1, 0, 1, 2, 3, 4, 5, 6, 7, 8, 9, 10, 11, 12, 13, 14, 15);
        acc = __builtin_amdgcn_wmma_f32_16x16x32_f16(
            false, af, false, bf, (short)0, acc, false, false);
    }

    const int rbase = (lane < 16) ? 0 : 8;
    if (n < 9) {
#pragma unroll
        for (int r = 0; r < 8; ++r)
            out[(size_t)(m0 + rbase + r) * 9 + n] = acc[r];
    }
}

// ---------------------------------------------------------------------------
// Host-side orchestration (all on `stream`, graph-capture safe).
// ---------------------------------------------------------------------------
extern "C" void kernel_launch(void* const* d_in, const int* in_sizes, int n_in,
                              void* d_out, int out_size, void* d_ws, size_t ws_size,
                              hipStream_t stream)
{
    const float* x  = (const float*)d_in[0];
    const float* W[6] = { (const float*)d_in[1], (const float*)d_in[3],
                          (const float*)d_in[5], (const float*)d_in[7],
                          (const float*)d_in[9], (const float*)d_in[11] };
    const float* b[6] = { (const float*)d_in[2], (const float*)d_in[4],
                          (const float*)d_in[6], (const float*)d_in[8],
                          (const float*)d_in[10], (const float*)d_in[12] };
    const float* g[5]  = { (const float*)d_in[13], (const float*)d_in[15],
                           (const float*)d_in[17], (const float*)d_in[19],
                           (const float*)d_in[21] };
    const float* be[5] = { (const float*)d_in[14], (const float*)d_in[16],
                           (const float*)d_in[18], (const float*)d_in[20],
                           (const float*)d_in[22] };

    const int B    = in_sizes[0] / 9;       // 262144
    const int nblk = B / 128;               // 2048 (B assumed multiple of 128)
    const float invB = 1.0f / (float)B;

    // Workspace layout
    char* ws = (char*)d_ws;
    _Float16* wf = (_Float16*)ws;                               // 143360 B of f16 weights
    float* P  = (float*)(ws + 147456);                          // [nblk][256] partials (2 MB)
    float* sc = (float*)(ws + 147456 + 2097152);                // [128]
    float* sh = sc + 128;                                       // [128]
    _Float16* hA = (_Float16*)(ws + 2248704);                   // [B][128] f16 (64 MB)
    _Float16* hB = hA + (size_t)B * 128;                        // [B][128] f16 (64 MB)

    _Float16* W1p = wf;                 // [128][32]  (K zero-padded)
    _Float16* W2f = wf + 4096;          // [128][128]
    _Float16* W3f = wf + 20480;
    _Float16* W4f = wf + 36864;
    _Float16* W5f = wf + 53248;
    _Float16* W6p = wf + 69632;         // [16][128]  (N zero-padded)

    prep_weights<<<280, 256, 0, stream>>>(W[0], W[1], W[2], W[3], W[4], W[5], wf);

    layer_first<<<nblk, 256, 0, stream>>>(x, W1p, b[0], hA, P);
    finalize_bn<<<1, 128, 0, stream>>>(P, nblk, g[0], be[0], sc, sh, invB);

    layer_mid<<<nblk, 256, 0, stream>>>(hA, sc, sh, W2f, b[1], hB, P);
    finalize_bn<<<1, 128, 0, stream>>>(P, nblk, g[1], be[1], sc, sh, invB);

    layer_mid<<<nblk, 256, 0, stream>>>(hB, sc, sh, W3f, b[2], hA, P);
    finalize_bn<<<1, 128, 0, stream>>>(P, nblk, g[2], be[2], sc, sh, invB);

    layer_mid<<<nblk, 256, 0, stream>>>(hA, sc, sh, W4f, b[3], hB, P);
    finalize_bn<<<1, 128, 0, stream>>>(P, nblk, g[3], be[3], sc, sh, invB);

    layer_mid<<<nblk, 256, 0, stream>>>(hB, sc, sh, W5f, b[4], hA, P);
    finalize_bn<<<1, 128, 0, stream>>>(P, nblk, g[4], be[4], sc, sh, invB);

    layer_last<<<nblk, 256, 0, stream>>>(hA, sc, sh, W6p, b[5], (float*)d_out);
}
